// InvSKAttention_2688649527316
// MI455X (gfx1250) — matrix operand order repaired and checked by reading, not compiled, not run on hardware
//
#include <hip/hip_runtime.h>
#include <hip/hip_fp16.h>

typedef __attribute__((ext_vector_type(16))) _Float16 v16h;
typedef __attribute__((ext_vector_type(8)))  float    v8f;

static constexpr int Bsz = 4, Csz = 512, HW = 1024, CR = 128;
#define BN_EPS 1e-5f

// ---------------------------------------------------------------------------
// prep: x[B,C,HW] f32 -> xT[B,HW,C] f16  (B-matrix operand for gemm1)
// ---------------------------------------------------------------------------
__global__ __launch_bounds__(256)
void k_prep_xT(const float* __restrict__ x, _Float16* __restrict__ xT) {
    int idx = blockIdx.x * 256 + threadIdx.x;          // B*HW*C = 2^21
    if (idx >= Bsz * HW * Csz) return;
    int b   = idx >> 19;                                // HW*C = 2^19
    int rem = idx & ((1 << 19) - 1);
    int p   = rem >> 9;                                 // C = 2^9
    int c   = rem & 511;
    xT[(b * HW + p) * Csz + c] = (_Float16)x[(b * Csz + c) * HW + p];
}

// ---------------------------------------------------------------------------
// prep: weights -> f16 (w1h [4][128][512], w2h padded [4][64][128]),
//       b2 padded [4][64] f32, Ssum zeroed
// ---------------------------------------------------------------------------
__global__ __launch_bounds__(256)
void k_prep_w(const float* __restrict__ w1, _Float16* __restrict__ w1h,
              const float* w2_1, const float* w2_3, const float* w2_5, const float* w2_7,
              const float* b2_1, const float* b2_3, const float* b2_5, const float* b2_7,
              _Float16* __restrict__ w2h, float* __restrict__ b2p,
              float* __restrict__ Ssum) {
    const int kk2[4] = {1, 9, 25, 49};
    const float* w2s[4] = {w2_1, w2_3, w2_5, w2_7};
    const float* b2s[4] = {b2_1, b2_3, b2_5, b2_7};
    int idx = blockIdx.x * 256 + threadIdx.x;
    if (idx < 4 * CR * Csz)                              // 262144
        w1h[idx] = (_Float16)w1[idx];
    if (idx < 4 * 64 * CR) {                             // 32768
        int i = idx >> 13, rem = idx & 8191;
        int m = rem >> 7, r = rem & 127;
        w2h[idx] = (m < kk2[i]) ? (_Float16)w2s[i][m * CR + r] : (_Float16)0.0f;
    }
    if (idx < 4 * 64) {
        int i = idx >> 6, m = idx & 63;
        b2p[idx] = (m < kk2[i]) ? b2s[i][m] : 0.0f;
    }
    if (idx < Bsz * Csz) Ssum[idx] = 0.0f;
}

// ---------------------------------------------------------------------------
// gemm1: t[br,b] = relu(BN(w1[br] @ x[b] + b1)) ; M=128 K=512 N=1024
// one wave per 16x16 tile, 8 waves cover all of M; grid = (Ntile, batch, br)
// output stored transposed: tT[br][b][HW][CR] f16
// ---------------------------------------------------------------------------
__global__ __launch_bounds__(256)
void k_gemm1(const _Float16* __restrict__ w1h, const _Float16* __restrict__ xT,
             const float* __restrict__ b1, const float* __restrict__ g1,
             const float* __restrict__ be1, _Float16* __restrict__ tT) {
    const int ntile = blockIdx.x, bb = blockIdx.y, br = blockIdx.z;
    const int wave = threadIdx.x >> 5;                   // = mtile 0..7
    const int lane = threadIdx.x & 31;
    const int half = lane >> 4, lrow = lane & 15;

    const _Float16* Arow = w1h + ((br * CR) + (wave * 16 + lrow)) * Csz;
    const _Float16* Brow = xT  + ((bb * HW) + (ntile * 16 + lrow)) * Csz;

    v8f acc = {0.f, 0.f, 0.f, 0.f, 0.f, 0.f, 0.f, 0.f};
#pragma unroll
    for (int k0 = 0; k0 < Csz; k0 += 32) {
        v16h a, b;
        // A (16x32 f16): lane holds row m=lrow, K-runs [k0+8*half,+8) and [k0+16+8*half,+8)
        ((float4*)&a)[0] = *(const float4*)(Arow + k0 + half * 8);
        ((float4*)&a)[1] = *(const float4*)(Arow + k0 + 16 + half * 8);
        // B (32x16 f16): lane holds col n=lrow, K-runs [k0+8*half,+8) and [k0+16+8*half,+8)
        ((float4*)&b)[0] = *(const float4*)(Brow + k0 + half * 8);
        ((float4*)&b)[1] = *(const float4*)(Brow + k0 + 16 + half * 8);
        acc = __builtin_amdgcn_wmma_f32_16x16x32_f16(false, a, false, b,
                                                     (short)0, acc, false, false);
    }
    // epilogue: bias + BN(scale,shift) + ReLU, store transposed row (8 contig f16)
    const int n = ntile * 16 + lrow;
    const float inv = rsqrtf(1.0f + BN_EPS);
    union { float4 f4; _Float16 h[8]; } u;
#pragma unroll
    for (int v = 0; v < 8; ++v) {
        int r = wave * 16 + half * 8 + v;                // C/D: M = v + 8*half
        float s = g1[br * CR + r] * inv;
        float t = (acc[v] + b1[br * CR + r]) * s + be1[br * CR + r];
        u.h[v] = (_Float16)fmaxf(t, 0.0f);
    }
    _Float16* dst = tT + (((br * Bsz + bb) * HW) + n) * CR + wave * 16 + half * 8;
    *(float4*)dst = u.f4;
}

// ---------------------------------------------------------------------------
// gemm2: wk[br,b] = w2[br] @ t + b2 ; M=K^2 (pad 16/16/32/64), K=128, N=1024
// ---------------------------------------------------------------------------
__global__ __launch_bounds__(128)
void k_gemm2(const _Float16* __restrict__ w2h, const _Float16* __restrict__ tT,
             const float* __restrict__ b2p, float* __restrict__ wk) {
    const int ntile = blockIdx.x, bb = blockIdx.y, br = blockIdx.z;
    const int wave = threadIdx.x >> 5;
    const int mtiles[4] = {1, 1, 2, 4};
    if (wave >= mtiles[br]) return;                      // wave-uniform exit
    const int lane = threadIdx.x & 31;
    const int half = lane >> 4, lrow = lane & 15;

    const _Float16* Arow = w2h + (br * 64 + wave * 16 + lrow) * CR;
    const _Float16* Brow = tT + (((br * Bsz + bb) * HW) + ntile * 16 + lrow) * CR;

    v8f acc = {0.f, 0.f, 0.f, 0.f, 0.f, 0.f, 0.f, 0.f};
#pragma unroll
    for (int k0 = 0; k0 < CR; k0 += 32) {
        v16h a, b;
        ((float4*)&a)[0] = *(const float4*)(Arow + k0 + half * 8);
        ((float4*)&a)[1] = *(const float4*)(Arow + k0 + 16 + half * 8);
        ((float4*)&b)[0] = *(const float4*)(Brow + k0 + half * 8);
        ((float4*)&b)[1] = *(const float4*)(Brow + k0 + 16 + half * 8);
        acc = __builtin_amdgcn_wmma_f32_16x16x32_f16(false, a, false, b,
                                                     (short)0, acc, false, false);
    }
    const int kk2[4] = {1, 9, 25, 49};
    const int n = ntile * 16 + lrow;
#pragma unroll
    for (int v = 0; v < 8; ++v) {
        int m = wave * 16 + half * 8 + v;
        if (m < kk2[br])
            wk[(((br * Bsz + bb) * 64) + m) * HW + n] = acc[v] + b2p[br * 64 + m];
    }
}

// ---------------------------------------------------------------------------
// involution apply + BN2 + ReLU -> feats ; partial pooled sum -> Ssum (atomic)
// grid = (b*c planes, branch); block = 256, 4 px per thread
// ---------------------------------------------------------------------------
__global__ __launch_bounds__(256)
void k_invol(const float* __restrict__ x, const float* __restrict__ wk,
             const float* __restrict__ g2, const float* __restrict__ be2,
             float* __restrict__ feats, float* __restrict__ Ssum) {
    const int br = blockIdx.y;
    const int bc = blockIdx.x;                           // b*512 + c
    const int bb = bc >> 9, c = bc & 511;
    const int ks[4] = {1, 3, 5, 7};
    const int k = ks[br], pad = (k - 1) >> 1;

    const float* xp  = x + (bb * Csz + c) * HW;
    const float* wkp = wk + ((br * Bsz + bb) * 64) * HW;
    const float  s  = g2[br * Csz + c] * rsqrtf(1.0f + BN_EPS);
    const float  sh = be2[br * Csz + c];

    float lsum = 0.0f;
    for (int p = threadIdx.x; p < HW; p += 256) {
        int h = p >> 5, w = p & 31;
        float acc = 0.0f;
        for (int kk = 0; kk < k * k; ++kk) {
            int hh = h + kk / k - pad;
            int ww = w + kk % k - pad;
            if ((unsigned)hh < 32u && (unsigned)ww < 32u)
                acc += xp[hh * 32 + ww] * wkp[kk * HW + p];
        }
        float f = fmaxf(acc * s + sh, 0.0f);
        feats[((br * Bsz + bb) * Csz + c) * HW + p] = f;
        lsum += f;
    }
    __shared__ float red[256];
    red[threadIdx.x] = lsum;
    __syncthreads();
    for (int off = 128; off > 0; off >>= 1) {
        if (threadIdx.x < off) red[threadIdx.x] += red[threadIdx.x + off];
        __syncthreads();
    }
    if (threadIdx.x == 0) atomicAdd(&Ssum[bb * Csz + c], red[0] * (1.0f / HW));
}

// ---------------------------------------------------------------------------
// Z[b,d] = S[b]·fc_w[d] + fc_b[d]   (tiny: 1 block, 128 threads)
// ---------------------------------------------------------------------------
__global__ void k_fc(const float* __restrict__ Ssum, const float* __restrict__ fcw,
                     const float* __restrict__ fcb, float* __restrict__ Z) {
    int t = threadIdx.x;
    if (t >= Bsz * 32) return;
    int b = t >> 5, d = t & 31;
    float z = fcb[d];
    for (int c = 0; c < Csz; ++c) z += Ssum[b * Csz + c] * fcw[d * Csz + c];
    Z[b * 32 + d] = z;
}

// ---------------------------------------------------------------------------
// logits + softmax over branches -> attn[4][b*512+c]
// ---------------------------------------------------------------------------
__global__ __launch_bounds__(256)
void k_attn(const float* __restrict__ Z, const float* __restrict__ fcsw,
            const float* __restrict__ fcsb, float* __restrict__ attn) {
    int idx = blockIdx.x * 256 + threadIdx.x;
    if (idx >= Bsz * Csz) return;
    int b = idx >> 9, c = idx & 511;
    float lg[4], mx = -1e30f;
    for (int i = 0; i < 4; ++i) {
        float v = fcsb[i * Csz + c];
        for (int d = 0; d < 32; ++d)
            v += Z[b * 32 + d] * fcsw[(i * Csz + c) * 32 + d];
        lg[i] = v;
        mx = fmaxf(mx, v);
    }
    float se = 0.0f;
    for (int i = 0; i < 4; ++i) { lg[i] = __expf(lg[i] - mx); se += lg[i]; }
    float inv = 1.0f / se;
    for (int i = 0; i < 4; ++i) attn[i * (Bsz * Csz) + idx] = lg[i] * inv;
}

// ---------------------------------------------------------------------------
// V = sum_i attn[i,b,c] * feats[i,b,c,:]   (float4 vectorized)
// ---------------------------------------------------------------------------
__global__ __launch_bounds__(256)
void k_fuse(const float* __restrict__ feats, const float* __restrict__ attn,
            float* __restrict__ out) {
    int bc = blockIdx.x;                                 // 0..2047
    int p4 = threadIdx.x;                                // 0..255 (HW/4)
    const float4* f4 = (const float4*)feats;
    float4 acc = {0.f, 0.f, 0.f, 0.f};
#pragma unroll
    for (int i = 0; i < 4; ++i) {
        float a = attn[i * 2048 + bc];
        float4 f = f4[(i * 2048 + bc) * 256 + p4];
        acc.x += a * f.x; acc.y += a * f.y; acc.z += a * f.z; acc.w += a * f.w;
    }
    ((float4*)out)[bc * 256 + p4] = acc;
}

// ---------------------------------------------------------------------------
extern "C" void kernel_launch(void* const* d_in, const int* in_sizes, int n_in,
                              void* d_out, int out_size, void* d_ws, size_t ws_size,
                              hipStream_t stream) {
    (void)in_sizes; (void)n_in; (void)out_size; (void)ws_size;
    const float* x     = (const float*)d_in[0];
    const float* w1    = (const float*)d_in[1];
    const float* b1    = (const float*)d_in[2];
    const float* bn1_g = (const float*)d_in[3];
    const float* bn1_b = (const float*)d_in[4];
    const float* bn2_g = (const float*)d_in[5];
    const float* bn2_b = (const float*)d_in[6];
    const float* fc_w  = (const float*)d_in[7];
    const float* fc_b  = (const float*)d_in[8];
    const float* fcs_w = (const float*)d_in[9];
    const float* fcs_b = (const float*)d_in[10];
    const float* w2_1  = (const float*)d_in[11];
    const float* b2_1  = (const float*)d_in[12];
    const float* w2_3  = (const float*)d_in[13];
    const float* b2_3  = (const float*)d_in[14];
    const float* w2_5  = (const float*)d_in[15];
    const float* b2_5  = (const float*)d_in[16];
    const float* w2_7  = (const float*)d_in[17];
    const float* b2_7  = (const float*)d_in[18];

    char* ws = (char*)d_ws;
    _Float16* xT    = (_Float16*)(ws + 0);               //  4 MiB  [B][HW][C]
    _Float16* w1h   = (_Float16*)(ws + 4194304);         //  512 KiB
    _Float16* w2h   = (_Float16*)(ws + 4718592);         //  64 KiB  [4][64][128]
    float*    b2p   = (float*)   (ws + 4784128);         //  1 KiB   [4][64]
    _Float16* tT    = (_Float16*)(ws + 4785152);         //  4 MiB  [4][B][HW][CR]
    float*    wk    = (float*)   (ws + 8979456);         //  4 MiB  [4][B][64][HW]
    float*    feats = (float*)   (ws + 13173760);        // 32 MiB  [4][B][C][HW]
    float*    Ssum  = (float*)   (ws + 46728192);        //  8 KiB  [B][C]
    float*    Z     = (float*)   (ws + 46736384);        //  [B][32]
    float*    attn  = (float*)   (ws + 46736896);        //  [4][B*C]

    k_prep_xT<<<dim3(8192), 256, 0, stream>>>(x, xT);
    k_prep_w <<<dim3(1024), 256, 0, stream>>>(w1, w1h, w2_1, w2_3, w2_5, w2_7,
                                              b2_1, b2_3, b2_5, b2_7, w2h, b2p, Ssum);
    k_gemm1  <<<dim3(64, 4, 4), 256, 0, stream>>>(w1h, xT, b1, bn1_g, bn1_b, tT);
    k_gemm2  <<<dim3(64, 4, 4), 128, 0, stream>>>(w2h, tT, b2p, wk);
    k_invol  <<<dim3(2048, 4), 256, 0, stream>>>(x, wk, bn2_g, bn2_b, feats, Ssum);
    k_fc     <<<dim3(1), 128, 0, stream>>>(Ssum, fc_w, fc_b, Z);
    k_attn   <<<dim3(8), 256, 0, stream>>>(Z, fcs_w, fcs_b, attn);
    k_fuse   <<<dim3(2048), 256, 0, stream>>>(feats, attn, d_out ? (float*)d_out : nullptr);
}
